// WindowBasedSelfAttention_58703613002177
// MI455X (gfx1250) — matrix operand order repaired and checked by reading, not compiled, not run on hardware
//
#include <hip/hip_runtime.h>
#include <hip/hip_bf16.h>
#include <math.h>

// ---------------------------------------------------------------------------
// CDNA5 (gfx1250) windowed Fourier self-attention.
// Heavy GEMMs: v_wmma_f32_16x16x32_f16, wave32, 16x32 output tile per wave.
// ---------------------------------------------------------------------------

typedef __attribute__((ext_vector_type(16))) _Float16 v16h;
typedef __attribute__((ext_vector_type(8)))  _Float16 v8h;
typedef __attribute__((ext_vector_type(8)))  float    v8f;

#define WMMA_F16(a, b, c) \
  __builtin_amdgcn_wmma_f32_16x16x32_f16(false, (a), false, (b), (short)0, (c), false, false)

// ----------------------------- constants -----------------------------------
#define NB    4          // batches
#define NWIN  961        // 31*31 windows per batch
#define NR    3844       // NB*NWIN sequence rows
#define NRP   3904       // padded rows (122*32), covers 3*961+992
#define DFULL 4096       // window spectrum size
#define DDIM  1024       // bottleneck dim
#define NP    992        // 961 padded to 31*32

// ----------------------- WMMA fragment loaders ------------------------------
// A fragment: 16x32 f16. lane lo = M row; elems[0..7]=K(hi*8+0..7),
// elems[8..15]=K(16+hi*8+0..7).   (ISA 7.12.2, 16-bit A-matrix)
__device__ inline v16h frag_a(const _Float16* A, long lda, long rowBase, int kb, int lane) {
  int lo = lane & 15, hi = lane >> 4;
  const _Float16* p = A + (rowBase + lo) * lda + kb + hi * 8;
  v8h x0 = *(const v8h*)p;
  v8h x1 = *(const v8h*)(p + 16);
  return __builtin_shufflevector(x0, x1, 0, 1, 2, 3, 4, 5, 6, 7, 8, 9, 10, 11, 12, 13, 14, 15);
}

// B fragment (NT: W stored N x K, K contiguous). lane lo = N col,
// elems[0..7] = W[n][kb + hi*8 + 0..7], elems[8..15] = W[n][kb + 16 + hi*8 + 0..7] : 2 x b128 loads.
__device__ inline v16h frag_b_nt(const _Float16* W, long ldw, long colBase, int kb, int lane) {
  int lo = lane & 15, hi = lane >> 4;
  const _Float16* p = W + (colBase + lo) * ldw + kb + hi * 8;
  v8h x0 = *(const v8h*)p;
  v8h x1 = *(const v8h*)(p + 16);
  return __builtin_shufflevector(x0, x1, 0, 1, 2, 3, 4, 5, 6, 7, 8, 9, 10, 11, 12, 13, 14, 15);
}

// ----------------------------- FFT helpers ----------------------------------
__device__ void fft64(float2* d, int s, float sign) {
  for (int i = 0; i < 64; i++) {                       // bit reversal (6 bits)
    int j = (int)(__brev((unsigned)i) >> 26);
    if (j > i) { float2 t = d[i * s]; d[i * s] = d[j * s]; d[j * s] = t; }
  }
  for (int len = 2; len <= 64; len <<= 1) {
    float ang = sign * 6.28318530717958647692f / (float)len;
    int half = len >> 1;
    for (int i = 0; i < 64; i += len) {
      for (int k = 0; k < half; k++) {
        float sw, cw;
        __sincosf(ang * (float)k, &sw, &cw);
        float2 u = d[(i + k) * s];
        float2 v = d[(i + k + half) * s];
        float tx = cw * v.x - sw * v.y;
        float ty = cw * v.y + sw * v.x;
        d[(i + k) * s]        = make_float2(u.x + tx, u.y + ty);
        d[(i + k + half) * s] = make_float2(u.x - tx, u.y - ty);
      }
    }
  }
}

// window extract + channel-mean + FFT2 + fftshift -> four (f32x2) + f16 planes
__global__ void k_fft_fwd(const float* __restrict__ x, float2* __restrict__ four,
                          _Float16* __restrict__ seqr, _Float16* __restrict__ seqi) {
  __shared__ float2 win[64][65];
  int w = blockIdx.x, b = blockIdx.y, t = threadIdx.x;
  int i = w / 31, j = w % 31;
  int gy = i * 32 + t;
  const float* p0 = x + ((long)(b * 3 + 0) * 1024 + gy) * 1024 + j * 32;
  const float* p1 = p0 + (long)1024 * 1024;
  const float* p2 = p1 + (long)1024 * 1024;
  for (int cc = 0; cc < 64; cc++) {
    float s = (p0[cc] + p1[cc] + p2[cc]) * (1.0f / 3.0f);
    win[t][cc] = make_float2(s, 0.f);
  }
  __syncthreads();
  fft64(&win[t][0], 1, -1.f);
  __syncthreads();
  fft64(&win[0][t], 65, -1.f);
  __syncthreads();
  long row = (long)b * NWIN + w;
  for (int cc = 0; cc < 64; cc++) {
    float2 v = win[(t + 32) & 63][(cc + 32) & 63];   // fftshift
    long idx = row * DFULL + t * 64 + cc;
    four[idx] = v;
    seqr[idx] = (_Float16)v.x;
    seqi[idx] = (_Float16)v.y;
  }
}

// ifftshift + IFFT2 -> real window
__global__ void k_fft_inv(const float2* __restrict__ spec, float* __restrict__ wout) {
  __shared__ float2 win[64][65];
  int w = blockIdx.x, b = blockIdx.y, t = threadIdx.x;
  long row = (long)b * NWIN + w;
  for (int cc = 0; cc < 64; cc++)
    win[t][cc] = spec[row * DFULL + (long)(((t + 32) & 63) * 64 + ((cc + 32) & 63))];
  __syncthreads();
  fft64(&win[t][0], 1, +1.f);
  __syncthreads();
  fft64(&win[0][t], 65, +1.f);
  __syncthreads();
  for (int cc = 0; cc < 64; cc++)
    wout[row * DFULL + t * 64 + cc] = win[t][cc].x * (1.0f / 4096.0f);
}

// complex64 -> two f16 planes
__global__ void k_c2planes(const float2* __restrict__ in, _Float16* __restrict__ re,
                           _Float16* __restrict__ im, int n) {
  int i = blockIdx.x * blockDim.x + threadIdx.x;
  if (i < n) {
    float2 v = in[i];
    re[i] = (_Float16)v.x;
    im[i] = (_Float16)v.y;
  }
}

// V (rows = b*961+m, cols d) -> vT [b][d][m], zero-padded m in [961,992)
__global__ void k_vtrans(const _Float16* __restrict__ vr, const _Float16* __restrict__ vi,
                         _Float16* __restrict__ vtr, _Float16* __restrict__ vti) {
  __shared__ _Float16 tr[32][33], ti[32][33];
  int b = blockIdx.z;
  int m0 = blockIdx.x * 32;   // 31 blocks -> 0..991
  int d0 = blockIdx.y * 32;   // 32 blocks -> 0..1023
  int tx = threadIdx.x & 31, ty = threadIdx.x >> 5;   // 32x8
  for (int r = ty; r < 32; r += 8) {
    int m = m0 + r, d = d0 + tx;
    _Float16 a = (_Float16)0.f, c = (_Float16)0.f;
    if (m < NWIN) {
      long o = ((long)b * NWIN + m) * DDIM + d;
      a = vr[o]; c = vi[o];
    }
    tr[r][tx] = a; ti[r][tx] = c;
  }
  __syncthreads();
  for (int r = ty; r < 32; r += 8) {
    int d = d0 + r, m = m0 + tx;
    long o = ((long)b * DDIM + d) * NP + m;
    vtr[o] = tr[tx][r]; vti[o] = ti[tx][r];
  }
}

// ----------------------------- complex GEMM (NT) -----------------------------
// Per wave: 16x32 complex output tile, 8 WMMAs / K-step.
enum { EPI_LRELU = 0, EPI_PLAIN = 1, EPI_ABS = 2, EPI_SIGFOUR = 3 };

template <int EPI>
__global__ void k_cgemm_nt(
    const _Float16* __restrict__ Ar, const _Float16* __restrict__ Ai, int lda, int aZRows,
    const _Float16* __restrict__ Wr, const _Float16* __restrict__ Wi, int ldw, int wZRows,
    const float2* __restrict__ bias,
    int Mvalid, int K, int mTiles, int nTiles2,
    _Float16* __restrict__ Or, _Float16* __restrict__ Oi, int ldo, int oZRows,
    float* __restrict__ Oabs, int ldabs, long oabsZ, float absScale,
    const float2* __restrict__ four, float2* __restrict__ spec) {
  int wave = threadIdx.x >> 5;
  int lane = threadIdx.x & 31;
  int tile = blockIdx.x * (blockDim.x >> 5) + wave;
  if (tile >= mTiles * nTiles2) return;
  int mT = tile / nTiles2, nT = tile % nTiles2;
  int z = blockIdx.y;
  long aRow = (long)z * aZRows + (long)mT * 16;
  long wRow = (long)z * wZRows + (long)nT * 32;

  v8f cr[2] = {}, cm[2] = {}, ci[2] = {};
  for (int kb = 0; kb < K; kb += 32) {
    if (kb + 32 < K) {   // CDNA5 global_prefetch of next A tiles
      __builtin_prefetch((const void*)(Ar + (aRow + (lane & 15)) * lda + kb + 32), 0, 3);
      __builtin_prefetch((const void*)(Ai + (aRow + (lane & 15)) * lda + kb + 32), 0, 3);
    }
    v16h ar = frag_a(Ar, lda, aRow, kb, lane);
    v16h ai = frag_a(Ai, lda, aRow, kb, lane);
#pragma unroll
    for (int ni = 0; ni < 2; ni++) {
      v16h br = frag_b_nt(Wr, ldw, wRow + ni * 16, kb, lane);
      v16h bi = frag_b_nt(Wi, ldw, wRow + ni * 16, kb, lane);
      cr[ni] = WMMA_F16(ar, br, cr[ni]);
      cm[ni] = WMMA_F16(ai, bi, cm[ni]);
      ci[ni] = WMMA_F16(ar, bi, ci[ni]);
      ci[ni] = WMMA_F16(ai, br, ci[ni]);
    }
  }

  int lo = lane & 15, hi = lane >> 4;
#pragma unroll
  for (int ni = 0; ni < 2; ni++) {
#pragma unroll
    for (int r = 0; r < 8; r++) {
      int m = mT * 16 + hi * 8 + r;
      int n = nT * 32 + ni * 16 + lo;
      float zr = cr[ni][r] - cm[ni][r];
      float zi = ci[ni][r];
      if (bias) { float2 bb = bias[n]; zr += bb.x; zi += bb.y; }
      if (EPI == EPI_LRELU) {
        zr = zr >= 0.f ? zr : 0.01f * zr;
        zi = zi >= 0.f ? zi : 0.01f * zi;
        if (m < Mvalid) {
          long o = ((long)z * oZRows + m) * ldo + n;
          Or[o] = (_Float16)zr; Oi[o] = (_Float16)zi;
        }
      } else if (EPI == EPI_PLAIN) {
        if (m < Mvalid) {
          long o = ((long)z * oZRows + m) * ldo + n;
          Or[o] = (_Float16)zr; Oi[o] = (_Float16)zi;
        }
      } else if (EPI == EPI_ABS) {
        Oabs[(long)z * oabsZ + (long)m * ldabs + n] = sqrtf(zr * zr + zi * zi) * absScale;
      } else {  // EPI_SIGFOUR: complex sigmoid gate * spectrum, in place
        if (m < Mvalid) {
          float e = __expf(-zr);
          float sv, cv;
          __sincosf(zi, &sv, &cv);
          float wx = 1.f + e * cv;
          float wy = -e * sv;
          float inv = 1.f / (wx * wx + wy * wy);
          float sgr = wx * inv, sgi = -wy * inv;   // 1/(1+exp(-z))
          long o = (long)m * DFULL + n;
          float2 f = four[o];
          spec[o] = make_float2(f.x * sgr - f.y * sgi, f.x * sgi + f.y * sgr);
        }
      }
    }
  }
}

// ----------------- real-A x complex-B GEMM (NT via vT) -----------------------
__global__ void k_rgemm_nt(
    const _Float16* __restrict__ A, int lda, int aZRows,            // attn [z][NP][NP]
    const _Float16* __restrict__ Wr, const _Float16* __restrict__ Wi, int ldw, int wZRows,
    int Mvalid, int K, int mTiles, int nTiles2,
    _Float16* __restrict__ Or, _Float16* __restrict__ Oi, int ldo, int oZRows) {
  int wave = threadIdx.x >> 5;
  int lane = threadIdx.x & 31;
  int tile = blockIdx.x * (blockDim.x >> 5) + wave;
  if (tile >= mTiles * nTiles2) return;
  int mT = tile / nTiles2, nT = tile % nTiles2;
  int z = blockIdx.y;
  long aRow = (long)z * aZRows + (long)mT * 16;
  long wRow = (long)z * wZRows + (long)nT * 32;

  v8f cr[2] = {}, ci[2] = {};
  for (int kb = 0; kb < K; kb += 32) {
    v16h a = frag_a(A, lda, aRow, kb, lane);
#pragma unroll
    for (int ni = 0; ni < 2; ni++) {
      v16h br = frag_b_nt(Wr, ldw, wRow + ni * 16, kb, lane);
      v16h bi = frag_b_nt(Wi, ldw, wRow + ni * 16, kb, lane);
      cr[ni] = WMMA_F16(a, br, cr[ni]);
      ci[ni] = WMMA_F16(a, bi, ci[ni]);
    }
  }
  int lo = lane & 15, hi = lane >> 4;
#pragma unroll
  for (int ni = 0; ni < 2; ni++) {
#pragma unroll
    for (int r = 0; r < 8; r++) {
      int m = mT * 16 + hi * 8 + r;
      int n = nT * 32 + ni * 16 + lo;
      if (m < Mvalid) {
        long o = ((long)z * oZRows + m) * ldo + n;
        Or[o] = (_Float16)cr[ni][r];
        Oi[o] = (_Float16)ci[ni][r];
      }
    }
  }
}

// ------------------------------- softmax -------------------------------------
__global__ void k_softmax(const float* __restrict__ S, _Float16* __restrict__ attn) {
  __shared__ float red[256];
  int r = blockIdx.x;   // 0..NP-1
  int b = blockIdx.y;
  int t = threadIdx.x;
  _Float16* arow = attn + ((long)b * NP + r) * NP;
  if (r >= NWIN) {      // padded rows -> zeros
    for (int m = t; m < NP; m += 256) arow[m] = (_Float16)0.f;
    return;
  }
  const float* srow = S + ((long)b * NP + r) * NP;
  float mx = -1e30f;
  for (int m = t; m < NWIN; m += 256) mx = fmaxf(mx, srow[m]);
  red[t] = mx; __syncthreads();
  for (int s = 128; s > 0; s >>= 1) { if (t < s) red[t] = fmaxf(red[t], red[t + s]); __syncthreads(); }
  mx = red[0]; __syncthreads();
  float sum = 0.f;
  for (int m = t; m < NWIN; m += 256) sum += __expf(srow[m] - mx);
  red[t] = sum; __syncthreads();
  for (int s = 128; s > 0; s >>= 1) { if (t < s) red[t] += red[t + s]; __syncthreads(); }
  float inv = 1.f / red[0];
  for (int m = t; m < NP; m += 256) {
    float v = (m < NWIN) ? __expf(srow[m] - mx) * inv : 0.f;
    arow[m] = (_Float16)v;
  }
}

// --------------------------- overlap merge -----------------------------------
__global__ void k_merge(const float* __restrict__ wout, float* __restrict__ out) {
  long idx = (long)blockIdx.x * blockDim.x + threadIdx.x;
  const long total = (long)NB * 1024 * 1024;
  if (idx >= total) return;
  int b = (int)(idx >> 20);
  int Y = (int)(idx >> 10) & 1023;
  int X = (int)idx & 1023;
  int ki = Y >> 6, u = Y & 63;
  int kj = X >> 6, w = X & 63;
  long cw = (long)b * NWIN + (2 * ki) * 31 + (2 * kj);   // corner window
  float val = wout[cw * DFULL + u * 64 + w];
  int ai = -1, aj = 0, au = 0, aw = 0;
  if (u >= 32 && w >= 32) { if (ki < 15 && kj < 15) { ai = ki;     aj = kj;     au = u - 32; aw = w - 32; } }
  else if (u >= 32)       { if (ki < 15 && kj >= 1) { ai = ki;     aj = kj - 1; au = u - 32; aw = w + 32; } }
  else if (w >= 32)       { if (ki >= 1 && kj < 15) { ai = ki - 1; aj = kj;     au = u + 32; aw = w - 32; } }
  else                    { if (ki >= 1 && kj >= 1) { ai = ki - 1; aj = kj - 1; au = u + 32; aw = w + 32; } }
  if (ai >= 0) {
    long an = (long)b * NWIN + (2 * ai + 1) * 31 + (2 * aj + 1);   // anchor window
    val = 0.5f * val + 0.5f * wout[an * DFULL + au * 64 + aw];
  }
  out[idx] = val;
}

// ------------------------------- launcher ------------------------------------
extern "C" void kernel_launch(void* const* d_in, const int* in_sizes, int n_in,
                              void* d_out, int out_size, void* d_ws, size_t ws_size,
                              hipStream_t stream) {
  (void)in_sizes; (void)n_in; (void)out_size; (void)ws_size;
  const float*  x      = (const float*)d_in[0];
  const float2* w_down = (const float2*)d_in[1];
  const float2* b_down = (const float2*)d_in[2];
  const float2* w_up   = (const float2*)d_in[3];
  const float2* b_up   = (const float2*)d_in[4];
  const float2* w_q    = (const float2*)d_in[5];
  const float2* b_q    = (const float2*)d_in[6];
  const float2* w_k    = (const float2*)d_in[7];
  const float2* b_k    = (const float2*)d_in[8];
  const float2* w_v    = (const float2*)d_in[9];
  const float2* b_v    = (const float2*)d_in[10];
  float* out = (float*)d_out;

  char* base = (char*)d_ws;
  size_t off = 0;
  auto alloc = [&](size_t bytes) -> void* {
    void* p = base + off;
    off = (off + bytes + 255) & ~(size_t)255;
    return p;
  };

  float2*   four = (float2*)  alloc((size_t)NR * DFULL * sizeof(float2));
  _Float16* seqr = (_Float16*)alloc((size_t)NRP * DFULL * 2);
  _Float16* seqi = (_Float16*)alloc((size_t)NRP * DFULL * 2);
  _Float16* wdr  = (_Float16*)alloc((size_t)DDIM * DFULL * 2);
  _Float16* wdi  = (_Float16*)alloc((size_t)DDIM * DFULL * 2);
  _Float16* wur  = (_Float16*)alloc((size_t)DFULL * DDIM * 2);
  _Float16* wui  = (_Float16*)alloc((size_t)DFULL * DDIM * 2);
  _Float16* wqr  = (_Float16*)alloc((size_t)DDIM * DDIM * 2);
  _Float16* wqi  = (_Float16*)alloc((size_t)DDIM * DDIM * 2);
  _Float16* wkr  = (_Float16*)alloc((size_t)DDIM * DDIM * 2);
  _Float16* wki  = (_Float16*)alloc((size_t)DDIM * DDIM * 2);
  _Float16* wvr  = (_Float16*)alloc((size_t)DDIM * DDIM * 2);
  _Float16* wvi  = (_Float16*)alloc((size_t)DDIM * DDIM * 2);
  _Float16* dr   = (_Float16*)alloc((size_t)NRP * DDIM * 2);
  _Float16* di   = (_Float16*)alloc((size_t)NRP * DDIM * 2);
  _Float16* qr   = (_Float16*)alloc((size_t)NRP * DDIM * 2);
  _Float16* qi   = (_Float16*)alloc((size_t)NRP * DDIM * 2);
  _Float16* kr   = (_Float16*)alloc((size_t)NRP * DDIM * 2);
  _Float16* ki   = (_Float16*)alloc((size_t)NRP * DDIM * 2);
  _Float16* vr   = (_Float16*)alloc((size_t)NRP * DDIM * 2);
  _Float16* vi   = (_Float16*)alloc((size_t)NRP * DDIM * 2);
  _Float16* vtr  = (_Float16*)alloc((size_t)NB * DDIM * NP * 2);
  _Float16* vti  = (_Float16*)alloc((size_t)NB * DDIM * NP * 2);
  float*    scr  = (float*)   alloc((size_t)NB * NP * NP * 4);
  _Float16* attn = (_Float16*)alloc((size_t)NB * NP * NP * 2);
  _Float16* wtr  = (_Float16*)alloc((size_t)NRP * DDIM * 2);
  _Float16* wti  = (_Float16*)alloc((size_t)NRP * DDIM * 2);
  float* wout = (float*)seqr;   // real IFFT windows alias the dead f16 seq planes

  // 1) windows -> spectrum
  k_fft_fwd<<<dim3(NWIN, NB), 64, 0, stream>>>(x, four, seqr, seqi);

  // 2) pack weights
  int n;
  n = DDIM * DFULL; k_c2planes<<<(n + 255) / 256, 256, 0, stream>>>(w_down, wdr, wdi, n);
  n = DFULL * DDIM; k_c2planes<<<(n + 255) / 256, 256, 0, stream>>>(w_up,   wur, wui, n);
  n = DDIM * DDIM;  k_c2planes<<<(n + 255) / 256, 256, 0, stream>>>(w_q,    wqr, wqi, n);
  n = DDIM * DDIM;  k_c2planes<<<(n + 255) / 256, 256, 0, stream>>>(w_k,    wkr, wki, n);
  n = DDIM * DDIM;  k_c2planes<<<(n + 255) / 256, 256, 0, stream>>>(w_v,    wvr, wvi, n);

  // 3) down-projection + complex leaky-ReLU
  {
    int mT = 241, nT2 = 32, tiles = mT * nT2;
    k_cgemm_nt<EPI_LRELU><<<dim3((tiles + 7) / 8, 1), 256, 0, stream>>>(
        seqr, seqi, DFULL, 0, wdr, wdi, DFULL, 0, b_down,
        NR, DFULL, mT, nT2, dr, di, DDIM, 0,
        nullptr, 0, 0, 0.f, nullptr, nullptr);
  }
  // 4) Q, K, V projections
  {
    int mT = 241, nT2 = 32, tiles = mT * nT2, blocks = (tiles + 7) / 8;
    k_cgemm_nt<EPI_PLAIN><<<dim3(blocks, 1), 256, 0, stream>>>(
        dr, di, DDIM, 0, wqr, wqi, DDIM, 0, b_q, NR, DDIM, mT, nT2,
        qr, qi, DDIM, 0, nullptr, 0, 0, 0.f, nullptr, nullptr);
    k_cgemm_nt<EPI_PLAIN><<<dim3(blocks, 1), 256, 0, stream>>>(
        dr, di, DDIM, 0, wkr, wki, DDIM, 0, b_k, NR, DDIM, mT, nT2,
        kr, ki, DDIM, 0, nullptr, 0, 0, 0.f, nullptr, nullptr);
    k_cgemm_nt<EPI_PLAIN><<<dim3(blocks, 1), 256, 0, stream>>>(
        dr, di, DDIM, 0, wvr, wvi, DDIM, 0, b_v, NR, DDIM, mT, nT2,
        vr, vi, DDIM, 0, nullptr, 0, 0, 0.f, nullptr, nullptr);
  }
  // 5) transpose V per batch -> vT [b][d][m], zero-padded
  k_vtrans<<<dim3(NP / 32, DDIM / 32, NB), 256, 0, stream>>>(vr, vi, vtr, vti);

  // 6) scores: per batch q x k^T, |.|/32 -> f32
  {
    int mT = NP / 16, nT2 = NP / 32, tiles = mT * nT2;
    k_cgemm_nt<EPI_ABS><<<dim3((tiles + 7) / 8, NB), 256, 0, stream>>>(
        qr, qi, DDIM, NWIN, kr, ki, DDIM, NWIN, nullptr,
        NP, DDIM, mT, nT2, nullptr, nullptr, 0, 0,
        scr, NP, (long)NP * NP, 0.03125f, nullptr, nullptr);
  }
  // 7) softmax over |scores| (zero-padded to NP cols)
  k_softmax<<<dim3(NP, NB), 256, 0, stream>>>(scr, attn);

  // 8) weighted = attn (real) x V via vT (NT), per batch
  {
    int mT = NP / 16, nT2 = DDIM / 32, tiles = mT * nT2;
    k_rgemm_nt<<<dim3((tiles + 7) / 8, NB), 256, 0, stream>>>(
        attn, NP, NP, vtr, vti, NP, DDIM,
        NWIN, NP, mT, nT2, wtr, wti, DDIM, NWIN);
  }
  // 9) up-projection + bias + complex sigmoid gate x spectrum (in place on four)
  {
    int mT = 241, nT2 = 128, tiles = mT * nT2;
    k_cgemm_nt<EPI_SIGFOUR><<<dim3((tiles + 7) / 8, 1), 256, 0, stream>>>(
        wtr, wti, DDIM, 0, wur, wui, DDIM, 0, b_up,
        NR, DDIM, mT, nT2, nullptr, nullptr, 0, 0,
        nullptr, 0, 0, 0.f, four, four);
  }
  // 10) inverse FFT -> real windows
  k_fft_inv<<<dim3(NWIN, NB), 64, 0, stream>>>(four, wout);

  // 11) overlap merge -> (4,1,1024,1024)
  {
    long total = (long)NB * 1024 * 1024;
    k_merge<<<dim3((unsigned)((total + 255) / 256)), 256, 0, stream>>>(wout, out);
  }
}
